// BPANSequential_89189290869294
// MI455X (gfx1250) — matrix operand order repaired and checked by reading, not compiled, not run on hardware
//
#include <hip/hip_runtime.h>
#include <hip/hip_bf16.h>

typedef _Float16 v16h __attribute__((ext_vector_type(16)));
typedef _Float16 v8h  __attribute__((ext_vector_type(8)));
typedef float    v8f  __attribute__((ext_vector_type(8)));

#define B_  256
#define L_  512
#define D_  128
#define NE_ 128
#define NI_ 32
#define C_  10
#define DT_ 0.2f

// workspace offsets (in halves)
#define OFF_XE 0
#define OFF_EE 16384
#define OFF_IE 32768
#define OFF_XI 36864
#define OFF_EI 40960
#define OFF_II 45056
#define OFF_RO 46080
#define OFF_XA 65536          // pre-packed x A-fragments: 16*512*2048 halves = 32 MB

// output offsets (floats): (logits_last, logits_seq, r_e_seq, r_i_seq, bal_e_seq, bal_i_seq)
#define O_LAST ((size_t)0)
#define O_LOG  ((size_t)2560)
#define O_RE   ((size_t)1313280)
#define O_RI   ((size_t)18090496)
#define O_BALE ((size_t)22284800)
#define O_BALI ((size_t)39062016)

#define RP  136   // LDS row stride (halves) for r_e buffer (272B -> 4-bank rotation per row)
#define RPI 40    // LDS row stride (halves) for r_i buffer

// ---------------------------------------------------------------------------
// Pack K x Nsrc row-major fp32 weight into WMMA 32x16 f16 B-fragment tiles:
// dst[((nt*KT+kt)*32 + lane)*16 + j] = w[kt*32 + (lane>>4)*8 + (j<8 ? j : j+8)][nt*16 + (lane&15)]
// Optionally |w| (Dale's law) and negation (inhibitory sign folded into weight).
// ---------------------------------------------------------------------------
__global__ void pack_weights(const float* __restrict__ src, unsigned short* __restrict__ dstu,
                             int Nsrc, int KT, int total, int doabs, int neg) {
  int idx = blockIdx.x * blockDim.x + threadIdx.x;
  if (idx >= total) return;
  _Float16* dst = reinterpret_cast<_Float16*>(dstu);
  int tile = idx >> 9;
  int r    = idx & 511;
  int lane = r >> 4;
  int j    = r & 15;
  int nt = tile / KT;
  int kt = tile - nt * KT;
  int k = kt * 32 + ((lane >> 4) << 3) + (j < 8 ? j : j + 8);
  int n = nt * 16 + (lane & 15);
  float v = (n < Nsrc) ? src[k * Nsrc + n] : 0.0f;   // pads W_ro cols 10..15 with 0
  if (doabs) v = fabsf(v);
  if (neg)   v = -v;
  dst[idx] = (_Float16)v;
}

// ---------------------------------------------------------------------------
// Pre-pack x_seq [B,L,D] fp32 into 16x32 f16 A-fragments:
//   xA[(bt*L+t)*2048 + kt*512 + lane*16 + j]
//     = x[bt*16 + (lane&15)][t][kt*32 + (lane>>4)*8 + (j<8 ? j : j+8)]
// ---------------------------------------------------------------------------
__global__ void pack_x(const float* __restrict__ x, unsigned short* __restrict__ dstu) {
  int idx = blockIdx.x * blockDim.x + threadIdx.x;   // 0 .. 2^21-1
  _Float16* xA = reinterpret_cast<_Float16*>(dstu) + OFF_XA;
  const int half = idx & 1;
  const int lane = (idx >> 1) & 31;
  const int kt   = (idx >> 6) & 3;
  const int bttt = idx >> 8;           // bt*L + t, 0..8191
  const int m    = lane & 15;
  const int t    = bttt & (L_ - 1);
  const int bt   = bttt >> 9;
  const int K0   = kt * 32 + (lane >> 4) * 8 + (half ? 16 : 0);
  const float* p = x + ((size_t)(bt * 16 + m) * L_ + t) * D_ + K0;
  const float4* q = (const float4*)p;
  float4 f0 = q[0], f1 = q[1];
  v8h h;
  h[0]=(_Float16)f0.x; h[1]=(_Float16)f0.y; h[2]=(_Float16)f0.z; h[3]=(_Float16)f0.w;
  h[4]=(_Float16)f1.x; h[5]=(_Float16)f1.y; h[6]=(_Float16)f1.z; h[7]=(_Float16)f1.w;
  *(v8h*)(xA + (size_t)bttt * 2048 + kt * 512 + lane * 16 + half * 8) = h;
}

// ---------------------------------------------------------------------------
// Fragment helpers
// ---------------------------------------------------------------------------
__device__ __forceinline__ v8f wmma16(v16h a, v16h b, v8f c) {
  return __builtin_amdgcn_wmma_f32_16x16x32_f16(false, a, false, b, (short)0, c, false, false);
}

__device__ __forceinline__ v16h lds_frag(const _Float16* p) {
  v8h lo = *(const v8h*)p;
  v8h hi = *(const v8h*)(p + 16);
  return __builtin_shufflevector(lo, hi, 0,1,2,3,4,5,6,7,8,9,10,11,12,13,14,15);
}

// ---------------------------------------------------------------------------
// Persistent E/I RNN kernel: 16 workgroups x 11 wave32s.
//   waves 0..7 : bal_e N-tiles (9 WMMAs/step), e-state in registers
//   waves 8..9 : bal_i N-tiles (9 WMMAs/step), i-state in registers
//   wave 10    : readout GEMM (4 WMMAs/step), pipelined one step behind
// One s_barrier per step; relu state ping-pongs through LDS. Stores use
// running base pointers with (r,t)-affine immediate offsets. x fragments for
// step t+1 are prefetched into registers during step t's MMA chain.
// ---------------------------------------------------------------------------
__global__ __launch_bounds__(352) void ei_rnn(const unsigned short* __restrict__ wsu,
                                              const float* __restrict__ b_e,
                                              const float* __restrict__ b_i,
                                              const float* __restrict__ b_ro,
                                              float* __restrict__ out) {
  const _Float16* wsB = reinterpret_cast<const _Float16*>(wsu);
  __shared__ __align__(16) _Float16 rbuf[2][16 * RP];
  __shared__ __align__(16) _Float16 ribuf[2][16 * RPI];

  const int tid  = threadIdx.x;
  const int wave = tid >> 5;          // wave32
  const int lane = tid & 31;
  const int m    = lane & 15;         // A-fragment row
  const int hk   = lane >> 4;         // half-of-K selector
  const int nl   = lane & 15;         // B/C/D column within tile
  const int bt   = blockIdx.x;        // batch tile 0..15

  // zero initial relu-state buffers
  {
    _Float16* p0 = &rbuf[0][0];
    for (int k2 = tid; k2 < 2 * 16 * RP; k2 += 352) p0[k2] = (_Float16)0.0f;
    _Float16* p1 = &ribuf[0][0];
    for (int k2 = tid; k2 < 2 * 16 * RPI; k2 += 352) p1[k2] = (_Float16)0.0f;
  }

  // preload this wave's B-fragments (invariant across all 512 steps)
  v16h Bw0 = {}, Bw1 = {}, Bw2 = {}, Bw3 = {}, Bw4 = {}, Bw5 = {}, Bw6 = {}, Bw7 = {}, Bw8 = {};
  float bias = 0.0f;
  v8f st = {};  // persistent e/i state tile in registers

  // hoisted base pointers (affine addressing; advanced per step)
  float* pbal = nullptr;
  float* prr  = nullptr;
  float* plog = nullptr;
  _Float16 *wa0 = nullptr, *wa1 = nullptr;   // LDS state write bases (ping/pong)

  // LDS A-fragment read bases (ping/pong), same for every role
  const _Float16* ra0 = &rbuf[0][m * RP + hk * 8];
  const _Float16* ra1 = &rbuf[1][m * RP + hk * 8];
  const _Float16* ia0 = &ribuf[0][m * RPI + hk * 8];
  const _Float16* ia1 = &ribuf[1][m * RPI + hk * 8];

  if (wave < 8) {
    const _Float16* px = wsB + OFF_XE + wave * 4 * 512 + lane * 16;
    const _Float16* pr = wsB + OFF_EE + wave * 4 * 512 + lane * 16;
    Bw0 = *(const v16h*)(px + 0 * 512); Bw1 = *(const v16h*)(px + 1 * 512);
    Bw2 = *(const v16h*)(px + 2 * 512); Bw3 = *(const v16h*)(px + 3 * 512);
    Bw4 = *(const v16h*)(pr + 0 * 512); Bw5 = *(const v16h*)(pr + 1 * 512);
    Bw6 = *(const v16h*)(pr + 2 * 512); Bw7 = *(const v16h*)(pr + 3 * 512);
    Bw8 = *(const v16h*)(wsB + OFF_IE + wave * 512 + lane * 16);
    bias = b_e[wave * 16 + nl];
    const int n = wave * 16 + nl;
    const size_t base = (size_t)(bt * 16 + hk * 8) * (L_ * NE_) + n;
    pbal = out + O_BALE + base;
    prr  = out + O_RE   + base;
    wa0 = &rbuf[0][hk * 8 * RP + n];
    wa1 = &rbuf[1][hk * 8 * RP + n];
  } else if (wave < 10) {
    const int wi = wave - 8;
    const _Float16* px = wsB + OFF_XI + wi * 4 * 512 + lane * 16;
    const _Float16* pr = wsB + OFF_EI + wi * 4 * 512 + lane * 16;
    Bw0 = *(const v16h*)(px + 0 * 512); Bw1 = *(const v16h*)(px + 1 * 512);
    Bw2 = *(const v16h*)(px + 2 * 512); Bw3 = *(const v16h*)(px + 3 * 512);
    Bw4 = *(const v16h*)(pr + 0 * 512); Bw5 = *(const v16h*)(pr + 1 * 512);
    Bw6 = *(const v16h*)(pr + 2 * 512); Bw7 = *(const v16h*)(pr + 3 * 512);
    Bw8 = *(const v16h*)(wsB + OFF_II + wi * 512 + lane * 16);
    bias = b_i[wi * 16 + nl];
    const int n = wi * 16 + nl;
    const size_t base = (size_t)(bt * 16 + hk * 8) * (L_ * NI_) + n;
    pbal = out + O_BALI + base;
    prr  = out + O_RI   + base;
    wa0 = &ribuf[0][hk * 8 * RPI + n];
    wa1 = &ribuf[1][hk * 8 * RPI + n];
  } else {
    const _Float16* pr = wsB + OFF_RO + lane * 16;
    Bw0 = *(const v16h*)(pr + 0 * 512); Bw1 = *(const v16h*)(pr + 1 * 512);
    Bw2 = *(const v16h*)(pr + 2 * 512); Bw3 = *(const v16h*)(pr + 3 * 512);
    bias = (nl < C_) ? b_ro[nl] : 0.0f;
    plog = out + O_LOG + (size_t)(bt * 16 + hk * 8) * (L_ * C_) + nl;
  }
  __syncthreads();

  // running pointer into this batch-tile's pre-packed x fragments + register
  // prefetch of step-0 fragments (E/I waves only)
  const _Float16* xq = wsB + OFF_XA + (size_t)bt * L_ * 2048 + lane * 16;
  v16h nx0 = {}, nx1 = {}, nx2 = {}, nx3 = {};
  if (wave < 10) {
    nx0 = *(const v16h*)(xq +    0);
    nx1 = *(const v16h*)(xq +  512);
    nx2 = *(const v16h*)(xq + 1024);
    nx3 = *(const v16h*)(xq + 1536);
  }

  for (int t = 0; t <= L_; ++t) {
    if (wave < 8) {
      if (t < L_) {
        // recurrent-state fragments: batch all ds_loads up front
        const _Float16* rb = (t & 1) ? ra1 : ra0;
        const v16h r0 = lds_frag(rb +  0);
        const v16h r1 = lds_frag(rb + 32);
        const v16h r2 = lds_frag(rb + 64);
        const v16h r3 = lds_frag(rb + 96);
        const v16h ri = lds_frag((t & 1) ? ia1 : ia0);
        v8f c = {};
        c = wmma16(nx0, Bw0, c);
        c = wmma16(nx1, Bw1, c);
        c = wmma16(nx2, Bw2, c);
        c = wmma16(nx3, Bw3, c);
        // x fragments consumed: start register-prefetch for step t+1 so the
        // global-load latency hides behind this step's MMAs + barrier
        if (t + 1 < L_) {
          xq += 2048;
          nx0 = *(const v16h*)(xq +    0);
          nx1 = *(const v16h*)(xq +  512);
          nx2 = *(const v16h*)(xq + 1024);
          nx3 = *(const v16h*)(xq + 1536);
          __builtin_prefetch(xq + 2048, 0, 3);   // warm L2 one further step out
        }
        c = wmma16(r0, Bw4, c);
        c = wmma16(r1, Bw5, c);
        c = wmma16(r2, Bw6, c);
        c = wmma16(r3, Bw7, c);
        c = wmma16(ri, Bw8, c);
        _Float16* wp = (t & 1) ? wa0 : wa1;   // write buffer (t+1)&1
#pragma unroll
        for (int r = 0; r < 8; ++r) {
          const float bal = c[r];
          pbal[r * (L_ * NE_)] = bal;                       // imm offset r*256KB
          st[r] = (1.0f - DT_) * st[r] + DT_ * (bal + bias);
          const float rv = fmaxf(st[r], 0.0f);
          prr[r * (L_ * NE_)] = rv;
          wp[r * RP] = (_Float16)rv;                        // imm offset r*272B
        }
        pbal += NE_; prr += NE_;
      }
    } else if (wave < 10) {
      if (t < L_) {
        const _Float16* rb = (t & 1) ? ra1 : ra0;
        const v16h r0 = lds_frag(rb +  0);
        const v16h r1 = lds_frag(rb + 32);
        const v16h r2 = lds_frag(rb + 64);
        const v16h r3 = lds_frag(rb + 96);
        const v16h ri = lds_frag((t & 1) ? ia1 : ia0);
        v8f c = {};
        c = wmma16(nx0, Bw0, c);
        c = wmma16(nx1, Bw1, c);
        c = wmma16(nx2, Bw2, c);
        c = wmma16(nx3, Bw3, c);
        if (t + 1 < L_) {
          xq += 2048;
          nx0 = *(const v16h*)(xq +    0);
          nx1 = *(const v16h*)(xq +  512);
          nx2 = *(const v16h*)(xq + 1024);
          nx3 = *(const v16h*)(xq + 1536);
        }
        c = wmma16(r0, Bw4, c);
        c = wmma16(r1, Bw5, c);
        c = wmma16(r2, Bw6, c);
        c = wmma16(r3, Bw7, c);
        c = wmma16(ri, Bw8, c);
        _Float16* wp = (t & 1) ? wa0 : wa1;
#pragma unroll
        for (int r = 0; r < 8; ++r) {
          const float bal = c[r];
          pbal[r * (L_ * NI_)] = bal;                       // imm offset r*64KB
          st[r] = (1.0f - DT_) * st[r] + DT_ * (bal + bias);
          const float rv = fmaxf(st[r], 0.0f);
          prr[r * (L_ * NI_)] = rv;
          wp[r * RPI] = (_Float16)rv;                       // imm offset r*80B
        }
        pbal += NI_; prr += NI_;
      }
    } else {
      if (t >= 1) {
        // logits for step t-1 from the relu-state written at end of step t-1
        const _Float16* rb = (t & 1) ? ra1 : ra0;
        const v16h r0 = lds_frag(rb +  0);
        const v16h r1 = lds_frag(rb + 32);
        const v16h r2 = lds_frag(rb + 64);
        const v16h r3 = lds_frag(rb + 96);
        v8f c = {};
        c = wmma16(r0, Bw0, c);
        c = wmma16(r1, Bw1, c);
        c = wmma16(r2, Bw2, c);
        c = wmma16(r3, Bw3, c);
        if (nl < C_) {
#pragma unroll
          for (int r = 0; r < 8; ++r) {
            const float v = c[r] + bias;
            plog[r * (L_ * C_)] = v;                        // imm offset r*20KB
            if (t == L_) out[O_LAST + (size_t)(bt * 16 + r + hk * 8) * C_ + nl] = v;
          }
        }
        plog += C_;
      }
    }
    __syncthreads();
  }
}

extern "C" void kernel_launch(void* const* d_in, const int* in_sizes, int n_in,
                              void* d_out, int out_size, void* d_ws, size_t ws_size,
                              hipStream_t stream) {
  (void)in_sizes; (void)n_in; (void)out_size; (void)ws_size;
  const float* x   = (const float*)d_in[0];
  const float* Wxe = (const float*)d_in[1];
  const float* Wxi = (const float*)d_in[2];
  const float* Wee = (const float*)d_in[3];
  const float* Wei = (const float*)d_in[4];
  const float* Wie = (const float*)d_in[5];
  const float* Wii = (const float*)d_in[6];
  const float* be  = (const float*)d_in[7];
  const float* bi  = (const float*)d_in[8];
  const float* Wro = (const float*)d_in[9];
  const float* bro = (const float*)d_in[10];
  unsigned short* ws = (unsigned short*)d_ws;   // 128KB weights + 32MB packed x

  auto pack = [&](const float* src, int off, int Nsrc, int NT, int KT, int doabs, int neg) {
    int total = NT * KT * 512;
    pack_weights<<<dim3((total + 255) / 256), dim3(256), 0, stream>>>(src, ws + off, Nsrc, KT,
                                                                      total, doabs, neg);
  };
  pack(Wxe, OFF_XE, 128, 8, 4, 0, 0);   // W_xe        128x128
  pack(Wee, OFF_EE, 128, 8, 4, 1, 0);   // |W_ee|      128x128
  pack(Wie, OFF_IE, 128, 8, 1, 1, 1);   // -|W_ie|      32x128
  pack(Wxi, OFF_XI,  32, 2, 4, 0, 0);   // W_xi        128x32
  pack(Wei, OFF_EI,  32, 2, 4, 1, 0);   // |W_ei|      128x32
  pack(Wii, OFF_II,  32, 2, 1, 1, 1);   // -|W_ii|      32x32
  pack(Wro, OFF_RO,  10, 1, 4, 0, 0);   // W_ro padded 128x16

  // pre-pack all of x_seq into f16 A-fragments (parallel over B*L, off critical path)
  pack_x<<<dim3((B_ / 16) * L_ * 4 * 32 * 2 / 256), dim3(256), 0, stream>>>(x, ws);

  ei_rnn<<<dim3(B_ / 16), dim3(352), 0, stream>>>(ws, be, bi, bro, (float*)d_out);
}
